// CapsuleRoutingSelfAttention_66022237274172
// MI455X (gfx1250) — matrix operand order repaired and checked by reading, not compiled, not run on hardware
//
#include <hip/hip_runtime.h>
#include <hip/hip_bf16.h>

// ---------------------------------------------------------------------------
// CDNA5 (gfx1250) capsule-routing self-attention.
// GEMMs: v_wmma_f32_16x16x32_bf16, LDS double-buffered staging through the
// ASYNCcnt-tracked global_load_async_to_lds path when available.
// ---------------------------------------------------------------------------

typedef __attribute__((ext_vector_type(16))) __bf16 v16bf;
typedef __attribute__((ext_vector_type(8)))  float  v8f;
typedef __attribute__((__vector_size__(16))) int    v4i;
typedef __hip_bfloat16 bf16_t;

union FragAB { v16bf v; uint4 q[2]; };
union FragC  { v8f   v; float f[8]; };

constexpr int Bc = 4, Hc = 8, Sc = 1024, Ec = 512, Dc = 64;
constexpr int HD = Hc * Dc;   // 512
constexpr int HK = Hc * Sc;   // 8192

#if __has_builtin(__builtin_amdgcn_global_load_async_to_lds_b128) && \
    __has_builtin(__builtin_amdgcn_s_wait_asynccnt)
#define USE_ASYNC_LDS 1
#else
#define USE_ASYNC_LDS 0
#endif

// 16-byte global -> LDS copy. Async (ASYNCcnt) when the toolchain exposes the
// gfx1250 builtin; otherwise plain load+store (compiler inserts waits).
__device__ __forceinline__ void copy16_g2l(const bf16_t* g, bf16_t* l)
{
#if USE_ASYNC_LDS
    __builtin_amdgcn_global_load_async_to_lds_b128(
        (__attribute__((address_space(1))) v4i*)(g),
        (__attribute__((address_space(3))) v4i*)(l), 0, 0);
#else
    *(uint4*)l = *(const uint4*)g;
#endif
}

// ---------------------------------------------------------------------------
// Batched NT GEMM:  C[z](M,N) = alpha * A[z](M,K) * B[z](N,K)^T (+ bias[n]) (+C)
// Block: 256 threads = 8 waves, block tile 128(M) x 64(N), K step 32.
// A/B tiles staged to LDS (rows padded to 40 bf16 -> conflict-free b128 reads),
// double buffered; each wave computes a 2x2 grid of 16x16 WMMA tiles.
// All problem dims used here are multiples of the tile sizes.
// ---------------------------------------------------------------------------
__global__ __launch_bounds__(256)
void gemm_bf16_nt(const bf16_t* __restrict__ A, const bf16_t* __restrict__ B,
                  float* __restrict__ C,
                  int K, int lda, int ldb, int ldc,
                  long long aS1, long long aS2,
                  long long bS1, long long bS2,
                  long long cS1, long long cS2, int zdiv,
                  float alpha, const float* __restrict__ bias, int accum)
{
    __shared__ __align__(16) bf16_t Asm[2][128][40];
    __shared__ __align__(16) bf16_t Bsm[2][64][40];

    const int z  = blockIdx.z;
    const int z1 = z / zdiv, z2 = z % zdiv;
    const bf16_t* Ab = A + (long long)z1 * aS1 + (long long)z2 * aS2;
    const bf16_t* Bb = B + (long long)z1 * bS1 + (long long)z2 * bS2;
    float*        Cb = C + (long long)z1 * cS1 + (long long)z2 * cS2;

    const int tid    = threadIdx.x;
    const int lane   = tid & 31;
    const int wid    = tid >> 5;
    const int blockM = blockIdx.y * 128;
    const int blockN = blockIdx.x * 64;
    const int m0w    = (wid & 3) * 32;   // wave tile origin inside block tile
    const int n0w    = (wid >> 2) * 32;
    const int la     = lane & 15;
    const int ks     = lane >> 4;

    // Per-thread staging assignment: A = 512 16B chunks (2/thread), B = 256.
    const int cr = tid >> 2;             // 0..63
    const int cc = (tid & 3) * 8;        // element offset of 16B chunk

    auto stage = [&](int buf, int k0) {
        const bf16_t* gA0 = Ab + (long long)(blockM + cr)      * lda + k0 + cc;
        const bf16_t* gA1 = Ab + (long long)(blockM + cr + 64) * lda + k0 + cc;
        const bf16_t* gB  = Bb + (long long)(blockN + cr)      * ldb + k0 + cc;
        copy16_g2l(gA0, &Asm[buf][cr][cc]);
        copy16_g2l(gA1, &Asm[buf][cr + 64][cc]);
        copy16_g2l(gB,  &Bsm[buf][cr][cc]);
        // Speculative warm of the k-tile after next (safe past end: dropped).
        __builtin_prefetch((const void*)(gA0 + 64), 0, 2);
        __builtin_prefetch((const void*)(gB  + 64), 0, 2);
    };

    FragC acc[2][2];
    #pragma unroll
    for (int i = 0; i < 2; ++i)
        #pragma unroll
        for (int j = 0; j < 2; ++j)
            #pragma unroll
            for (int r = 0; r < 8; ++r) acc[i][j].f[r] = 0.0f;

    const int nk = K >> 5;
    stage(0, 0);
    for (int kt = 0; kt < nk; ++kt) {
        if (kt + 1 < nk) stage((kt + 1) & 1, (kt + 1) << 5);
#if USE_ASYNC_LDS
        if (kt + 1 < nk) __builtin_amdgcn_s_wait_asynccnt(3); // prev stage done
        else             __builtin_amdgcn_s_wait_asynccnt(0);
#endif
        __syncthreads();

        const int buf = kt & 1;
        FragAB afr[2], bfr[2];
        // A frag (16x32): lane<16 -> K {0..7,16..23}, lane>=16 -> {8..15,24..31}
        #pragma unroll
        for (int i = 0; i < 2; ++i) {
            const bf16_t* pa = &Asm[buf][m0w + i*16 + la][ks * 8];
            afr[i].q[0] = *(const uint4*)(pa);
            afr[i].q[1] = *(const uint4*)(pa + 16);
        }
        // B frag (32x16 NT): lane<16 -> K {0..7,16..23}, lane>=16 -> {8..15,24..31}
        #pragma unroll
        for (int j = 0; j < 2; ++j) {
            const bf16_t* pb = &Bsm[buf][n0w + j*16 + la][ks * 8];
            bfr[j].q[0] = *(const uint4*)(pb);
            bfr[j].q[1] = *(const uint4*)(pb + 16);
        }
        #pragma unroll
        for (int i = 0; i < 2; ++i)
            #pragma unroll
            for (int j = 0; j < 2; ++j)
                acc[i][j].v = __builtin_amdgcn_wmma_f32_16x16x32_bf16(
                    false, afr[i].v, false, bfr[j].v,
                    (short)0, acc[i][j].v, false, false);
        __syncthreads();   // before next stage overwrites this buffer
    }

    // C layout: VGPR r -> M = r (lanes 0-15) / r+8 (lanes 16-31); N = lane&15.
    const int rbase = ks * 8;
    #pragma unroll
    for (int i = 0; i < 2; ++i) {
        #pragma unroll
        for (int j = 0; j < 2; ++j) {
            const int col  = blockN + n0w + j*16 + la;
            const float bv = bias ? bias[col] : 0.0f;
            #pragma unroll
            for (int r = 0; r < 8; ++r) {
                const int row = blockM + m0w + i*16 + rbase + r;
                const long long off = (long long)row * ldc + col;
                float v = acc[i][j].f[r] * alpha + bv;
                if (accum) v += Cb[off];
                Cb[off] = v;
            }
        }
    }
}

// ---------------------------------------------------------------------------
// Elementwise fp32 -> bf16
// ---------------------------------------------------------------------------
__global__ void f32_to_bf16_kernel(const float* __restrict__ in,
                                   bf16_t* __restrict__ out, long long n)
{
    long long t = (long long)blockIdx.x * blockDim.x + threadIdx.x;
    if (t < n) out[t] = __float2bfloat16(in[t]);
}

// ---------------------------------------------------------------------------
// Batched transpose + convert: out[z][c][r] = (bf16) in[z][r][c]
// ---------------------------------------------------------------------------
__global__ void transpose_f32_to_bf16(const float* __restrict__ in,
                                      bf16_t* __restrict__ out,
                                      int M, int N, int ldin, int ldout,
                                      long long inB1, long long inB2,
                                      long long outB1, long long outB2, int zdiv)
{
    const int z = blockIdx.y, z1 = z / zdiv, z2 = z % zdiv;
    long long t = (long long)blockIdx.x * blockDim.x + threadIdx.x;
    if (t >= (long long)M * N) return;
    const int r = (int)(t / N), c = (int)(t % N);
    out[(long long)z1*outB1 + (long long)z2*outB2 + (long long)c*ldout + r] =
        __float2bfloat16(in[(long long)z1*inB1 + (long long)z2*inB2 +
                            (long long)r*ldin + c]);
}

// ---------------------------------------------------------------------------
// Build token-capsule matrix u[b,j,h*S+k] and its transpose uT[b,h*S+k,j]
// from scores[b,h,j,k] (bf16 operands for the routing WMMA GEMMs).
// ---------------------------------------------------------------------------
__global__ void build_u_kernel(const float* __restrict__ scores,
                               bf16_t* __restrict__ u, bf16_t* __restrict__ uT)
{
    const long long t = (long long)blockIdx.x * 256 + threadIdx.x; // < 2^25
    const int k = (int)(t & 1023);
    const int j = (int)((t >> 10) & 1023);
    const int h = (int)((t >> 20) & 7);
    const int b = (int)(t >> 23);
    const bf16_t bv = __float2bfloat16(scores[t]);
    u [((long long)b * Sc + j) * HK + (long long)h * Sc + k] = bv;
    uT[((long long)b * HK + (long long)h * Sc + k) * Sc + j] = bv;
}

// ---------------------------------------------------------------------------
// Vertical (head-wise) capsule routing, fused per (b,q).
// 8 score rows (8x1024 f32 = 32KB) cached in LDS; 3 routing iterations.
// ---------------------------------------------------------------------------
__global__ __launch_bounds__(256)
void vertical_routing_kernel(const float* __restrict__ scores,
                             float* __restrict__ vout)
{
    __shared__ float rows[Hc][Sc];
    __shared__ float red[256];
    __shared__ float bvec[Hc];
    const int tid = threadIdx.x;
    const int b   = blockIdx.x >> 10;
    const int qi  = blockIdx.x & 1023;

    for (int t = tid; t < Hc * Sc; t += 256) {
        const int h = t >> 10, k = t & 1023;
        rows[h][k] = scores[((long long)(b * Hc + h) << 20) +
                            ((long long)qi << 10) + k];
    }
    if (tid < Hc) bvec[tid] = 0.0f;
    __syncthreads();

    const int k0 = tid * 4;
    float v0 = 0.f, v1 = 0.f, v2 = 0.f, v3 = 0.f;
    for (int iter = 0; iter < 3; ++iter) {
        float bm = bvec[0];
        #pragma unroll
        for (int h = 1; h < Hc; ++h) bm = fmaxf(bm, bvec[h]);
        float ch[Hc], cs = 0.f;
        #pragma unroll
        for (int h = 0; h < Hc; ++h) { ch[h] = __expf(bvec[h] - bm); cs += ch[h]; }
        const float inv = 1.0f / cs;

        float s0 = 0.f, s1 = 0.f, s2 = 0.f, s3 = 0.f;
        #pragma unroll
        for (int h = 0; h < Hc; ++h) {
            const float c = ch[h] * inv;
            s0 += c * rows[h][k0+0]; s1 += c * rows[h][k0+1];
            s2 += c * rows[h][k0+2]; s3 += c * rows[h][k0+3];
        }
        float ss = s0*s0 + s1*s1 + s2*s2 + s3*s3;
        red[tid] = ss; __syncthreads();
        for (int st = 128; st > 0; st >>= 1) {
            if (tid < st) red[tid] += red[tid + st];
            __syncthreads();
        }
        const float sq = red[0]; __syncthreads();
        const float f  = sq / ((1.0f + sq) * sqrtf(sq + 1e-8f));
        v0 = f*s0; v1 = f*s1; v2 = f*s2; v3 = f*s3;

        if (iter < 2) {  // agreement update b_h += <row_h, v>
            for (int h = 0; h < Hc; ++h) {
                float p = rows[h][k0+0]*v0 + rows[h][k0+1]*v1 +
                          rows[h][k0+2]*v2 + rows[h][k0+3]*v3;
                red[tid] = p; __syncthreads();
                for (int st = 128; st > 0; st >>= 1) {
                    if (tid < st) red[tid] += red[tid + st];
                    __syncthreads();
                }
                if (tid == 0) bvec[h] += red[0];
                __syncthreads();
            }
        }
    }
    float* o = vout + ((long long)(b * Sc + qi) << 10);
    o[k0+0] = v0; o[k0+1] = v1; o[k0+2] = v2; o[k0+3] = v3;
}

// ---------------------------------------------------------------------------
// Causal-masked row softmax of routing logits b[b,i,:] -> coupling c (bf16).
// ---------------------------------------------------------------------------
__global__ __launch_bounds__(256)
void causal_softmax_kernel(const float* __restrict__ bmat,
                           bf16_t* __restrict__ c)
{
    __shared__ float red[256];
    const int tid = threadIdx.x;
    const int b   = blockIdx.x >> 10;
    const int i   = blockIdx.x & 1023;
    const float* brow = bmat + ((long long)(b * Sc + i) << 10);
    bf16_t*      crow = c    + ((long long)(b * Sc + i) << 10);
    const int j0 = tid * 4;

    float x[4]; float m = -1e30f;
    #pragma unroll
    for (int kk = 0; kk < 4; ++kk) {
        const int j = j0 + kk;
        x[kk] = (j <= i) ? brow[j] : -1e9f;
        m = fmaxf(m, x[kk]);
    }
    red[tid] = m; __syncthreads();
    for (int st = 128; st > 0; st >>= 1) {
        if (tid < st) red[tid] = fmaxf(red[tid], red[tid + st]);
        __syncthreads();
    }
    const float gm = red[0]; __syncthreads();
    float s = 0.f;
    #pragma unroll
    for (int kk = 0; kk < 4; ++kk) { x[kk] = __expf(x[kk] - gm); s += x[kk]; }
    red[tid] = s; __syncthreads();
    for (int st = 128; st > 0; st >>= 1) {
        if (tid < st) red[tid] += red[tid + st];
        __syncthreads();
    }
    const float inv = 1.0f / red[0];
    #pragma unroll
    for (int kk = 0; kk < 4; ++kk)
        crow[j0 + kk] = __float2bfloat16(x[kk] * inv);
}

// ---------------------------------------------------------------------------
// squash() over rows of s[rows, HK]; writes v back in place (fp32) and a bf16
// copy for the next WMMA GEMM.
// ---------------------------------------------------------------------------
__global__ __launch_bounds__(256)
void squash_rows_kernel(float* __restrict__ s, bf16_t* __restrict__ vb)
{
    __shared__ float red[256];
    const int tid = threadIdx.x;
    float*  row = s  + (long long)blockIdx.x * HK;
    bf16_t* ob  = vb + (long long)blockIdx.x * HK;
    float x[32]; float ss = 0.f;
    #pragma unroll
    for (int ii = 0; ii < 32; ++ii) {
        x[ii] = row[tid + ii * 256];
        ss += x[ii] * x[ii];
    }
    red[tid] = ss; __syncthreads();
    for (int st = 128; st > 0; st >>= 1) {
        if (tid < st) red[tid] += red[tid + st];
        __syncthreads();
    }
    const float sq = red[0]; __syncthreads();
    const float f  = sq / ((1.0f + sq) * sqrtf(sq + 1e-8f));
    #pragma unroll
    for (int ii = 0; ii < 32; ++ii) {
        const float v = f * x[ii];
        row[tid + ii * 256] = v;
        ob [tid + ii * 256] = __float2bfloat16(v);
    }
}

// ---------------------------------------------------------------------------
// enhanced = scores + v_vert (bcast over h) + v_horiz; row softmax -> w (bf16)
// ---------------------------------------------------------------------------
__global__ __launch_bounds__(256)
void enhanced_softmax_kernel(const float* __restrict__ scores,
                             const float* __restrict__ vv,
                             const float* __restrict__ vh,
                             bf16_t* __restrict__ w)
{
    __shared__ float red[256];
    const int tid = threadIdx.x;
    const int z   = blockIdx.x;              // (b*H + h)*S + q
    const int b   = z >> 13;
    const int h   = (z >> 10) & 7;
    const int qi  = z & 1023;
    const float* srow = scores + ((long long)z << 10);
    const float* vvr  = vv + ((long long)(b * Sc + qi) << 10);
    const float* vhr  = vh + (long long)(b * Sc + qi) * HK + (long long)h * Sc;
    bf16_t* wrow = w + ((long long)z << 10);
    const int j0 = tid * 4;

    float x[4]; float m = -1e30f;
    #pragma unroll
    for (int kk = 0; kk < 4; ++kk) {
        const int j = j0 + kk;
        x[kk] = srow[j] + vvr[j] + vhr[j];
        m = fmaxf(m, x[kk]);
    }
    red[tid] = m; __syncthreads();
    for (int st = 128; st > 0; st >>= 1) {
        if (tid < st) red[tid] = fmaxf(red[tid], red[tid + st]);
        __syncthreads();
    }
    const float gm = red[0]; __syncthreads();
    float s = 0.f;
    #pragma unroll
    for (int kk = 0; kk < 4; ++kk) { x[kk] = __expf(x[kk] - gm); s += x[kk]; }
    red[tid] = s; __syncthreads();
    for (int st = 128; st > 0; st >>= 1) {
        if (tid < st) red[tid] += red[tid + st];
        __syncthreads();
    }
    const float inv = 1.0f / red[0];
    #pragma unroll
    for (int kk = 0; kk < 4; ++kk)
        wrow[j0 + kk] = __float2bfloat16(x[kk] * inv);
}

// ---------------------------------------------------------------------------
// Host-side orchestration
// ---------------------------------------------------------------------------
extern "C" void kernel_launch(void* const* d_in, const int* in_sizes, int n_in,
                              void* d_out, int out_size, void* d_ws, size_t ws_size,
                              hipStream_t stream)
{
    (void)in_sizes; (void)n_in; (void)out_size; (void)ws_size;
    const float* x  = (const float*)d_in[0];
    const float* Wq = (const float*)d_in[1];
    const float* bq = (const float*)d_in[2];
    const float* Wk = (const float*)d_in[3];
    const float* bk = (const float*)d_in[4];
    const float* Wv = (const float*)d_in[5];
    const float* bv = (const float*)d_in[6];
    const float* Wo = (const float*)d_in[7];
    const float* bo = (const float*)d_in[8];
    float* out = (float*)d_out;

    // ---- workspace carve-up (256B aligned) ----
    char* p = (char*)d_ws;
    auto alloc = [&](size_t bytes) -> void* {
        void* r = (void*)p; p += (bytes + 255) & ~(size_t)255; return r;
    };
    const long long BS   = (long long)Bc * Sc;           // 4096
    const long long NBHS = (long long)Bc * Hc * Sc * Sc; // 2^25

    bf16_t* xb    = (bf16_t*)alloc(BS * Ec * 2);
    bf16_t* WqT   = (bf16_t*)alloc((size_t)Ec * HD * 2);
    bf16_t* WkT   = (bf16_t*)alloc((size_t)Ec * HD * 2);
    bf16_t* WvT   = (bf16_t*)alloc((size_t)Ec * HD * 2);
    bf16_t* WoT   = (bf16_t*)alloc((size_t)HD * Ec * 2);
    float*  qf    = (float*) alloc(BS * HD * 4);
    float*  kf    = (float*) alloc(BS * HD * 4);
    float*  vf    = (float*) alloc(BS * HD * 4);
    bf16_t* qb    = (bf16_t*)alloc(BS * HD * 2);
    bf16_t* kb    = (bf16_t*)alloc(BS * HD * 2);
    float*  sc    = (float*) alloc(NBHS * 4);            // scores
    bf16_t* u     = (bf16_t*)alloc(NBHS * 2);
    bf16_t* uT    = (bf16_t*)alloc(NBHS * 2);
    float*  bhor  = (float*) alloc((size_t)Bc * Sc * Sc * 4);
    bf16_t* cb    = (bf16_t*)alloc((size_t)Bc * Sc * Sc * 2);
    float*  sbuf  = (float*) alloc(NBHS * 4);            // s, then v_horiz
    bf16_t* vhb   = (bf16_t*)alloc(NBHS * 2);
    float*  vvert = (float*) alloc((size_t)Bc * Sc * Sc * 4);
    bf16_t* wbuf  = (bf16_t*)alloc(NBHS * 2);
    bf16_t* vT    = (bf16_t*)alloc((size_t)Bc * Hc * Dc * Sc * 2);
    float*  ctx   = (float*) alloc(BS * HD * 4);
    bf16_t* ctxb  = (bf16_t*)alloc(BS * HD * 2);

    const dim3 blk(256);

    // 1. x -> bf16
    f32_to_bf16_kernel<<<dim3((BS*Ec)/256), blk, 0, stream>>>(x, xb, BS*Ec);
    // 2. W transposes (NT GEMM wants B as [N,K])
    transpose_f32_to_bf16<<<dim3((Ec*HD)/256, 1), blk, 0, stream>>>(
        Wq, WqT, Ec, HD, HD, Ec, 0, 0, 0, 0, 1);
    transpose_f32_to_bf16<<<dim3((Ec*HD)/256, 1), blk, 0, stream>>>(
        Wk, WkT, Ec, HD, HD, Ec, 0, 0, 0, 0, 1);
    transpose_f32_to_bf16<<<dim3((Ec*HD)/256, 1), blk, 0, stream>>>(
        Wv, WvT, Ec, HD, HD, Ec, 0, 0, 0, 0, 1);
    transpose_f32_to_bf16<<<dim3((HD*Ec)/256, 1), blk, 0, stream>>>(
        Wo, WoT, HD, Ec, Ec, HD, 0, 0, 0, 0, 1);

    // 3. QKV projections: [4096x512] x [512x512]^T
    gemm_bf16_nt<<<dim3(HD/64, BS/128, 1), blk, 0, stream>>>(
        xb, WqT, qf, Ec, Ec, Ec, HD, 0,0, 0,0, 0,0, 1, 1.0f, bq, 0);
    gemm_bf16_nt<<<dim3(HD/64, BS/128, 1), blk, 0, stream>>>(
        xb, WkT, kf, Ec, Ec, Ec, HD, 0,0, 0,0, 0,0, 1, 1.0f, bk, 0);
    gemm_bf16_nt<<<dim3(HD/64, BS/128, 1), blk, 0, stream>>>(
        xb, WvT, vf, Ec, Ec, Ec, HD, 0,0, 0,0, 0,0, 1, 1.0f, bv, 0);
    f32_to_bf16_kernel<<<dim3((BS*HD)/256), blk, 0, stream>>>(qf, qb, BS*HD);
    f32_to_bf16_kernel<<<dim3((BS*HD)/256), blk, 0, stream>>>(kf, kb, BS*HD);

    // 4. scores[b,h] = (1/8) q[b,h] k[b,h]^T, batched over 32 (b,h)
    gemm_bf16_nt<<<dim3(Sc/64, Sc/128, Bc*Hc), blk, 0, stream>>>(
        qb, kb, sc, Dc, HD, HD, Sc,
        (long long)Sc*HD, Dc,                  // A: b-stride, h-stride
        (long long)Sc*HD, Dc,                  // B
        (long long)Hc*Sc*Sc, (long long)Sc*Sc, // C
        Hc, 0.125f, nullptr, 0);

    // 5. vertical (head-wise) routing, fused
    vertical_routing_kernel<<<dim3(Bc*Sc), blk, 0, stream>>>(sc, vvert);

    // 6. horizontal routing operands u / uT (bf16)
    build_u_kernel<<<dim3((unsigned)(NBHS/256)), blk, 0, stream>>>(sc, u, uT);
    (void)hipMemsetAsync(bhor, 0, (size_t)Bc * Sc * Sc * 4, stream);

    // 7. three routing iterations (host-unrolled)
    for (int it = 0; it < 3; ++it) {
        causal_softmax_kernel<<<dim3(Bc*Sc), blk, 0, stream>>>(bhor, cb);
        // s[b] = c[b] (1024x1024) * uT[b] (8192x1024)^T  -> [1024x8192]
        gemm_bf16_nt<<<dim3(HK/64, Sc/128, Bc), blk, 0, stream>>>(
            cb, uT, sbuf, Sc, Sc, Sc, HK,
            (long long)Sc*Sc, 0, (long long)HK*Sc, 0, (long long)Sc*HK, 0,
            1, 1.0f, nullptr, 0);
        squash_rows_kernel<<<dim3(Bc*Sc), blk, 0, stream>>>(sbuf, vhb);
        if (it < 2) {
            // b[b] += v[b] (1024x8192) * u[b] (1024x8192)^T
            gemm_bf16_nt<<<dim3(Sc/64, Sc/128, Bc), blk, 0, stream>>>(
                vhb, u, bhor, HK, HK, HK, Sc,
                (long long)Sc*HK, 0, (long long)Sc*HK, 0, (long long)Sc*Sc, 0,
                1, 1.0f, nullptr, 1);
        }
    }

    // 8. enhanced = scores + v_vert + v_horiz; softmax -> w (bf16)
    enhanced_softmax_kernel<<<dim3(Bc*Hc*Sc), blk, 0, stream>>>(
        sc, vvert, sbuf, wbuf);

    // 9. v^T per (b,h): [1024x64] -> [64x1024] bf16
    transpose_f32_to_bf16<<<dim3((Sc*Dc)/256, Bc*Hc), blk, 0, stream>>>(
        vf, vT, Sc, Dc, HD, Sc,
        (long long)Sc*HD, Dc,                  // in:  b-stride, h-stride
        (long long)Hc*Dc*Sc, (long long)Dc*Sc, // out
        Hc);

    // 10. ctx[b,q,h*64+d] = sum_k w[b,h,q,k] v[b,k,h,d]
    gemm_bf16_nt<<<dim3(Dc/64, Sc/128, Bc*Hc), blk, 0, stream>>>(
        wbuf, vT, ctx, Sc, Sc, Sc, HD,
        (long long)Hc*Sc*Sc, (long long)Sc*Sc,
        (long long)Hc*Dc*Sc, (long long)Dc*Sc,
        (long long)Sc*HD, Dc,
        Hc, 1.0f, nullptr, 0);

    // 11. out = ctx @ Wo + bo : [4096x512] x [512x512]^T
    f32_to_bf16_kernel<<<dim3((BS*HD)/256), blk, 0, stream>>>(ctx, ctxb, BS*HD);
    gemm_bf16_nt<<<dim3(Ec/64, BS/128, 1), blk, 0, stream>>>(
        ctxb, WoT, out, HD, HD, HD, Ec, 0,0, 0,0, 0,0, 1, 1.0f, bo, 0);
}
